// GCNconv_two_aggregators_Net_67508295958856
// MI455X (gfx1250) — matrix operand order repaired and checked by reading, not compiled, not run on hardware
//
#include <hip/hip_runtime.h>
#include <hip/hip_bf16.h>

#define N_NODES 50000
#define N_EDGES 640000
#define DIM     128
#define N_GRAPHS 128

typedef __attribute__((ext_vector_type(16))) __bf16 v16bf;
typedef __attribute__((ext_vector_type(8)))  float  v8f;

static __device__ __forceinline__ __bf16 f2bf(float f) {
  // native fp32 -> bf16 convert (gfx1250 has hardware bf16 cvt; avoids
  // the 3-VALU-per-element manual RNE emulation)
  return static_cast<__bf16>(f);
}

// ---------------------------------------------------------------------------
// GEMM: C[M x 128] (+)= A[M x 128] * B[128 x 128]  via v_wmma_f32_16x16x32_bf16
// flags: bit0 = accumulate into existing C, bit1 = ReLU epilogue
// One wave computes one 16x16 tile; 4 K-steps of 32.
// ---------------------------------------------------------------------------
__global__ void __launch_bounds__(128)
gemm_nk128_bf16_wmma(const float* __restrict__ A, const float* __restrict__ B,
                     const float* __restrict__ bias, float* __restrict__ C,
                     int M, int flags) {
  const int lane = threadIdx.x & 31;
  const int wave = threadIdx.x >> 5;
  const int tiles = (M >> 4) * (DIM >> 4);
  int t = blockIdx.x * 4 + wave;
  if (t >= tiles) return;
  const int tm = t >> 3;      // 8 column tiles (DIM/16)
  const int tn = t & 7;
  const int g  = lane >> 4;   // half-wave group
  const int mn = lane & 15;   // row (for A) / col (for B,D) within tile

  const float* arow = A + (size_t)(tm * 16 + mn) * DIM;
  const int col0 = tn * 16;

  v8f acc;
  if (flags & 1) {
#pragma unroll
    for (int r = 0; r < 8; ++r)
      acc[r] = C[(size_t)(tm * 16 + r + 8 * g) * DIM + col0 + mn];
  } else {
#pragma unroll
    for (int r = 0; r < 8; ++r) acc[r] = 0.f;
  }

#pragma unroll
  for (int kt = 0; kt < 4; ++kt) {
    const int kbase = kt * 32;
    v16bf a, b;
    // A lane layout (16-bit A 16x32): g=0 -> K {0..7,16..23}, g=1 -> K {8..15,24..31}
    const float4* ap0 = (const float4*)(arow + kbase + 8 * g);
    const float4* ap1 = (const float4*)(arow + kbase + 16 + 8 * g);
    float4 a0 = ap0[0], a1 = ap0[1], a2 = ap1[0], a3 = ap1[1];
    a[0]=f2bf(a0.x);  a[1]=f2bf(a0.y);  a[2]=f2bf(a0.z);  a[3]=f2bf(a0.w);
    a[4]=f2bf(a1.x);  a[5]=f2bf(a1.y);  a[6]=f2bf(a1.z);  a[7]=f2bf(a1.w);
    a[8]=f2bf(a2.x);  a[9]=f2bf(a2.y);  a[10]=f2bf(a2.z); a[11]=f2bf(a2.w);
    a[12]=f2bf(a3.x); a[13]=f2bf(a3.y); a[14]=f2bf(a3.z); a[15]=f2bf(a3.w);
    // B lane layout (32x16): g=0 -> K {0..7,16..23}, g=1 -> K {8..15,24..31}
#pragma unroll
    for (int i = 0; i < 16; ++i)
      b[i] = f2bf(B[(size_t)(kbase + 8 * g + (i & 7) + 16 * (i >> 3)) * DIM + col0 + mn]);
    acc = __builtin_amdgcn_wmma_f32_16x16x32_bf16(false, a, false, b,
                                                  (short)0, acc, false, false);
  }

  const float bb = bias ? bias[col0 + mn] : 0.f;
  const bool relu = (flags & 2) != 0;
#pragma unroll
  for (int r = 0; r < 8; ++r) {
    float v = acc[r] + bb;
    if (relu) v = fmaxf(v, 0.f);
    C[(size_t)(tm * 16 + r + 8 * g) * DIM + col0 + mn] = v;
  }
}

// ---------------------------------------------------------------------------
__global__ void edge_degree_kernel(const int* __restrict__ dst,
                                   float* __restrict__ deg, int E) {
  int e = blockIdx.x * blockDim.x + threadIdx.x;
  if (e < E) unsafeAtomicAdd(&deg[dst[e]], 1.0f);
}

__global__ void inv_sqrt_deg_kernel(float* __restrict__ deg, int n) {
  int i = blockIdx.x * blockDim.x + threadIdx.x;
  if (i < n) deg[i] = rsqrtf(deg[i] + 1.0f);
}

// one wave per edge: gather h[src] (float4/lane), scatter-add norm*h into out[dst]
__global__ void __launch_bounds__(256)
edge_aggregate_kernel(const int* __restrict__ src, const int* __restrict__ dst,
                      const float* __restrict__ dis, const float* __restrict__ h,
                      float* __restrict__ out, int E) {
  const int lane = threadIdx.x & 31;
  int e = blockIdx.x * 8 + (threadIdx.x >> 5);
  if (e >= E) return;
  int s = src[e], d = dst[e];
  float w = dis[s] * dis[d];
  float4 v = ((const float4*)(h + (size_t)s * DIM))[lane];
  float* op = out + (size_t)d * DIM + lane * 4;
  unsafeAtomicAdd(op + 0, w * v.x);
  unsafeAtomicAdd(op + 1, w * v.y);
  unsafeAtomicAdd(op + 2, w * v.z);
  unsafeAtomicAdd(op + 3, w * v.w);
}

// agg = relu(agg + dis^2 * h + bias)
__global__ void selfloop_bias_relu_kernel(float* __restrict__ agg,
                                          const float* __restrict__ h,
                                          const float* __restrict__ dis,
                                          const float* __restrict__ bias, int n) {
  int idx = blockIdx.x * blockDim.x + threadIdx.x;
  if (idx >= n * DIM) return;
  int i = idx >> 7, f = idx & 127;
  float s = dis[i];
  float v = agg[idx] + s * s * h[idx] + bias[f];
  agg[idx] = fmaxf(v, 0.f);
}

__global__ void __launch_bounds__(256)
pool_kernel(const float* __restrict__ H, const int* __restrict__ batch,
            float* __restrict__ pooled, int n) {
  const int lane = threadIdx.x & 31;
  int i = blockIdx.x * 8 + (threadIdx.x >> 5);
  if (i >= n) return;
  int b = batch[i];
  float4 v = ((const float4*)(H + (size_t)i * DIM))[lane];
  float* op = pooled + (size_t)b * DIM + lane * 4;
  unsafeAtomicAdd(op + 0, v.x);
  unsafeAtomicAdd(op + 1, v.y);
  unsafeAtomicAdd(op + 2, v.z);
  unsafeAtomicAdd(op + 3, v.w);
}

__global__ void final_linear_kernel(const float* __restrict__ pooled,
                                    const float* __restrict__ Wl,
                                    const float* __restrict__ bl,
                                    float* __restrict__ out) {
  int gidx = threadIdx.x;                 // 128 graphs, one thread each
  const float* p = pooled + (size_t)gidx * DIM;
  float s = bl[0];
#pragma unroll 4
  for (int f = 0; f < DIM; ++f) s += p[f] * Wl[f];
  out[gidx] = s;
}

// ---------------------------------------------------------------------------
extern "C" void kernel_launch(void* const* d_in, const int* in_sizes, int n_in,
                              void* d_out, int out_size, void* d_ws, size_t ws_size,
                              hipStream_t stream) {
  (void)in_sizes; (void)n_in; (void)out_size; (void)ws_size;
  const float* x    = (const float*)d_in[0];
  const int*   eil  = (const int*)d_in[1];   // [2,E] local
  const int*   eig  = (const int*)d_in[2];   // [2,E] global
  const int*   batch= (const int*)d_in[3];
  const float* W11=(const float*)d_in[4];  const float* b11=(const float*)d_in[5];
  const float* W12=(const float*)d_in[6];  const float* b12=(const float*)d_in[7];
  const float* M1a=(const float*)d_in[8];  const float* m1a=(const float*)d_in[9];
  const float* M1b=(const float*)d_in[10]; const float* m1b=(const float*)d_in[11];
  const float* W21=(const float*)d_in[12]; const float* b21=(const float*)d_in[13];
  const float* W22=(const float*)d_in[14]; const float* b22=(const float*)d_in[15];
  const float* M2a=(const float*)d_in[16]; const float* m2a=(const float*)d_in[17];
  const float* M2b=(const float*)d_in[18]; const float* m2b=(const float*)d_in[19];
  const float* Wl =(const float*)d_in[20]; const float* bl =(const float*)d_in[21];
  float* out = (float*)d_out;

  // workspace carve-up (fp32)
  const size_t NM = (size_t)N_NODES * DIM;
  float* ws   = (float*)d_ws;
  float* HA   = ws;               // [N,128] x@W (local path) / layer outputs
  float* HB   = HA + NM;          // [N,128] x@W (global path)
  float* AGA  = HB + NM;          // [N,128] aggregation (local)  -> x1
  float* AGB  = AGA + NM;         // [N,128] aggregation (global) -> x2
  float* T    = AGB + NM;         // [N,128] MLP hidden
  float* Hm   = T + NM;           // [N,128] MLP output h
  float* degl = Hm + NM;          // [N] local  deg -> dis
  float* degg = degl + N_NODES;   // [N] global deg -> dis
  float* pooled = degg + N_NODES; // [G,128]

  const int* srcL = eil;            const int* dstL = eil + N_EDGES;
  const int* srcG = eig;            const int* dstG = eig + N_EDGES;

  const int tiles   = (N_NODES / 16) * (DIM / 16);   // 25000
  const int gblocks = (tiles + 3) / 4;
  const int eblocks = (N_EDGES + 7) / 8;
  const int nblocks = (N_NODES + 7) / 8;
  const int sblocks = (N_NODES * DIM + 255) / 256;

  auto gemm = [&](const float* A, const float* B, const float* bias, float* Cc, int flags) {
    gemm_nk128_bf16_wmma<<<gblocks, 128, 0, stream>>>(A, B, bias, Cc, N_NODES, flags);
  };

  // degrees / normalization (shared by both layers)
  hipMemsetAsync(degl, 0, 2 * N_NODES * sizeof(float), stream);
  edge_degree_kernel<<<(N_EDGES + 255) / 256, 256, 0, stream>>>(dstL, degl, N_EDGES);
  edge_degree_kernel<<<(N_EDGES + 255) / 256, 256, 0, stream>>>(dstG, degg, N_EDGES);
  inv_sqrt_deg_kernel<<<(2 * N_NODES + 255) / 256, 256, 0, stream>>>(degl, 2 * N_NODES);

  // ---- layer 1 -------------------------------------------------------------
  gemm(x, W11, nullptr, HA, 0);
  gemm(x, W12, nullptr, HB, 0);
  hipMemsetAsync(AGA, 0, 2 * NM * sizeof(float), stream);     // AGA + AGB
  edge_aggregate_kernel<<<eblocks, 256, 0, stream>>>(srcL, dstL, degl, HA, AGA, N_EDGES);
  edge_aggregate_kernel<<<eblocks, 256, 0, stream>>>(srcG, dstG, degg, HB, AGB, N_EDGES);
  selfloop_bias_relu_kernel<<<sblocks, 256, 0, stream>>>(AGA, HA, degl, b11, N_NODES);
  selfloop_bias_relu_kernel<<<sblocks, 256, 0, stream>>>(AGB, HB, degg, b12, N_NODES);
  // mlp_1: T = relu(x1@M1a_top + x2@M1a_bot + m1a); Hm = T@M1b + m1b
  gemm(AGA, M1a, nullptr, T, 0);
  gemm(AGB, M1a + DIM * DIM, m1a, T, 1 | 2);
  gemm(T, M1b, m1b, Hm, 0);

  // ---- layer 2 -------------------------------------------------------------
  gemm(Hm, W21, nullptr, HA, 0);
  gemm(Hm, W22, nullptr, HB, 0);
  hipMemsetAsync(AGA, 0, 2 * NM * sizeof(float), stream);
  edge_aggregate_kernel<<<eblocks, 256, 0, stream>>>(srcL, dstL, degl, HA, AGA, N_EDGES);
  edge_aggregate_kernel<<<eblocks, 256, 0, stream>>>(srcG, dstG, degg, HB, AGB, N_EDGES);
  selfloop_bias_relu_kernel<<<sblocks, 256, 0, stream>>>(AGA, HA, degl, b21, N_NODES);
  selfloop_bias_relu_kernel<<<sblocks, 256, 0, stream>>>(AGB, HB, degg, b22, N_NODES);
  // mlp_2: T = relu(...); HA = T@M2b + m2b   (HA reused as final node features)
  gemm(AGA, M2a, nullptr, T, 0);
  gemm(AGB, M2a + DIM * DIM, m2a, T, 1 | 2);
  gemm(T, M2b, m2b, HA, 0);

  // ---- pooling + final linear ---------------------------------------------
  hipMemsetAsync(pooled, 0, (size_t)N_GRAPHS * DIM * sizeof(float), stream);
  pool_kernel<<<nblocks, 256, 0, stream>>>(HA, batch, pooled, N_NODES);
  final_linear_kernel<<<1, N_GRAPHS, 0, stream>>>(pooled, Wl, bl, out);
}
